// NormLinearAttention_1906965479820
// MI455X (gfx1250) — matrix operand order repaired and checked by reading, not compiled, not run on hardware
//
#include <hip/hip_runtime.h>
#include <hip/hip_bf16.h>
#include <math.h>

// ---------------- problem constants ----------------
#define BATCH 2
#define NSEQ  2048
#define DMODEL 1024
#define NHEAD 16
#define DHEAD 64          // per-head dim before LRPE
#define DLRPE 128         // 2*DHEAD after LRPE
#define ROWS  (BATCH * NSEQ)      // 4096
#define LN_EPS 1e-5f

// ---------------- types ----------------
typedef __bf16 bf16_t;
typedef __attribute__((ext_vector_type(16))) __bf16 v16bf;
typedef __attribute__((ext_vector_type(8)))  float  v8f;
typedef __attribute__((ext_vector_type(4)))  unsigned int u32x4;
typedef __attribute__((ext_vector_type(8)))  int i32x8;
typedef __attribute__((ext_vector_type(4)))  int i32x4;

#if defined(__has_builtin)
#if __has_builtin(__builtin_amdgcn_tensor_load_to_lds) && __has_builtin(__builtin_amdgcn_s_wait_tensorcnt)
#define HAVE_TDM 1
#endif
#endif
#ifndef HAVE_TDM
#define HAVE_TDM 0
#endif

__device__ __forceinline__ bf16_t f2bf(float f) {
    unsigned int u = __builtin_bit_cast(unsigned int, f);
    unsigned int r = (u + 0x7fffu + ((u >> 16) & 1u)) >> 16;   // RNE to bf16
    unsigned short s = (unsigned short)r;
    return __builtin_bit_cast(bf16_t, s);
}
__device__ __forceinline__ float bf2f(bf16_t h) {
    unsigned short s = __builtin_bit_cast(unsigned short, h);
    unsigned int u = ((unsigned int)s) << 16;
    return __builtin_bit_cast(float, u);
}
__device__ __forceinline__ float silu_f(float x) {
    return x / (1.0f + __expf(-x));
}

// ---------------- fp32 -> bf16 convert ----------------
__global__ __launch_bounds__(256)
void cvt_f32_bf16(const float* __restrict__ in, bf16_t* __restrict__ out, int n) {
    int i = blockIdx.x * 256 + threadIdx.x;
    if (i < n) out[i] = f2bf(in[i]);
}

// ---------------- generic bf16 WMMA GEMM ----------------
// C[M,N] = A[M,K] @ Bw[N,K]^T (+ bias). Block tile 128x128, staged K-tile 64
// (two WMMA K=32 substeps per tile => 16 v_wmma between barriers), 256 threads
// = 8 waves (4Mx2N), each wave 32x64. Tiles staged to LDS by the Tensor Data
// Mover (double-buffered, TENSORcnt-tracked) when available; sync loads else.
#define TKM 128
#define TKN 128
#define TKK 32            // WMMA K step
#define TILE_K 64         // staged K width per TDM/load tile
#define LDS_STR 72        // padded LDS row stride (144B): conflict-free b128 reads
#define LDS_STR_H 40      // head-GEMM stride (32-half rows + 16B pad)

#if HAVE_TDM
// Async 2D tile load (tile_w x nrows halves) global->LDS via TDM.
// D# packing per CDNA5 ISA. Hardware LDS row
// padding: pad_interval code 4 = 32 DWORDs (128B data) + pad_amount code 3
// = 4 DWORDs (16B) -> 144B LDS row stride = LDS_STR halves.
__device__ __forceinline__ void tdm_load_tile_bf16(const bf16_t* g, unsigned lds_off,
                                                   unsigned rowlen_halves, unsigned nrows,
                                                   unsigned tile_w_halves,
                                                   unsigned pad_i_code, unsigned pad_a_code) {
    unsigned long long ga = (unsigned long long)(uintptr_t)g;
    u32x4 g0;
    g0[0] = 1u;                                        // count=1: valid descriptor
    g0[1] = lds_off;                                   // LDS byte address (addr[31:0])
    g0[2] = (unsigned)ga;                              // global_addr[31:0]
    g0[3] = (unsigned)((ga >> 32) & 0x01FFFFFFull)     // global_addr[56:32]
          | 0x80000000u;                               // type=2 ("image")
    i32x8 g1;
    g1[0] = (int)((1u << 16)                           // data_size = 2 bytes
          | (1u << 20)                                 // pad_enable
          | (pad_i_code << 22)                         // DWORDs between pads
          | (pad_a_code << 25));                       // pad DWORDs
    g1[1] = (int)((rowlen_halves & 0xFFFFu) << 16);    // tensor_dim0[15:0]
    g1[2] = (int)(((rowlen_halves >> 16) & 0xFFFFu)    // tensor_dim0[31:16]
          | ((nrows & 0xFFFFu) << 16));                // tensor_dim1[15:0]
    g1[3] = (int)(((nrows >> 16) & 0xFFFFu)            // tensor_dim1[31:16]
          | (tile_w_halves << 16));                    // tile_dim0
    g1[4] = (int)(nrows & 0xFFFFu);                    // tile_dim1 rows, tile_dim2=0
    g1[5] = (int)rowlen_halves;                        // tensor_dim0_stride[31:0]
    g1[6] = 0;                                         // stride0[47:32], stride1[15:0]
    g1[7] = 0;
    i32x4 z4 = {0, 0, 0, 0};                           // groups 2/3 unused (2D tensor)
#if __clang_major__ >= 23
    i32x8 z8 = {0, 0, 0, 0, 0, 0, 0, 0};
    __builtin_amdgcn_tensor_load_to_lds(g0, g1, z4, z4, z8, 0);
#else
    __builtin_amdgcn_tensor_load_to_lds(g0, g1, z4, z4, 0);
#endif
}
#endif

__global__ __launch_bounds__(256)
void gemm_bf16_wmma(const bf16_t* __restrict__ A,
                    const bf16_t* __restrict__ Bw,
                    const float* __restrict__ bias,
                    float* __restrict__ C,
                    int M, int N, int K) {
    __shared__ __align__(16) bf16_t As[2][TKM * LDS_STR];
    __shared__ __align__(16) bf16_t Bs[2][TKN * LDS_STR];

    const int tid  = threadIdx.x;
    const int wave = tid >> 5;
    const int lane = tid & 31;
    const int wm   = wave >> 1;         // 0..3
    const int wn   = wave & 1;          // 0..1
    const int rowBase = blockIdx.y * TKM;
    const int colBase = blockIdx.x * TKN;
    const int rsel = lane & 15;
    const int hsel = lane >> 4;         // 0 or 1

    v8f acc[2][4];
    #pragma unroll
    for (int mi = 0; mi < 2; ++mi)
        #pragma unroll
        for (int ni = 0; ni < 4; ++ni)
            #pragma unroll
            for (int j = 0; j < 8; ++j) acc[mi][ni][j] = 0.0f;

    int cur = 0;
#if HAVE_TDM
    const bool issuer = (tid < 32);     // wave 0 owns the TDM queue
    const bf16_t* Abase = A  + (size_t)rowBase * K;
    const bf16_t* Bbase = Bw + (size_t)colBase * K;
    const unsigned ldsA[2] = { (unsigned)(uintptr_t)&As[0][0], (unsigned)(uintptr_t)&As[1][0] };
    const unsigned ldsB[2] = { (unsigned)(uintptr_t)&Bs[0][0], (unsigned)(uintptr_t)&Bs[1][0] };
    if (issuer) {                       // prologue: tile 0 in flight
        tdm_load_tile_bf16(Abase, ldsA[0], (unsigned)K, TKM, TILE_K, 4, 3);
        tdm_load_tile_bf16(Bbase, ldsB[0], (unsigned)K, TKN, TILE_K, 4, 3);
    }
#endif

    for (int k0 = 0; k0 < K; k0 += TILE_K) {
#if HAVE_TDM
        const bool more = (k0 + TILE_K) < K;
        if (issuer) {
            if (more) {                 // issue tile k0+TILE_K into the other buffer
                tdm_load_tile_bf16(Abase + k0 + TILE_K, ldsA[cur ^ 1], (unsigned)K, TKM, TILE_K, 4, 3);
                tdm_load_tile_bf16(Bbase + k0 + TILE_K, ldsB[cur ^ 1], (unsigned)K, TKN, TILE_K, 4, 3);
                __builtin_amdgcn_s_wait_tensorcnt(2);   // drain tile k0, keep next in flight
            } else {
                __builtin_amdgcn_s_wait_tensorcnt(0);
            }
        }
        __syncthreads();                // tile k0 visible to all 8 waves
#else
        // synchronous staging: 128x64 halves each side; 1024 b128 chunks; 4/thread
        #pragma unroll
        for (int i = 0; i < 4; ++i) {
            int idx = tid + i * 256;
            int r = idx >> 3;
            int c = (idx & 7) * 8;
            *(uint4*)&As[0][r * LDS_STR + c] =
                *(const uint4*)&A[(size_t)(rowBase + r) * K + k0 + c];
            *(uint4*)&Bs[0][r * LDS_STR + c] =
                *(const uint4*)&Bw[(size_t)(colBase + r) * K + k0 + c];
        }
        __syncthreads();
        if (k0 + TILE_K < K) {          // global_prefetch_b8 of next K-tiles
            __builtin_prefetch(&A[(size_t)(rowBase + (tid >> 1)) * K + k0 + TILE_K + (tid & 1) * 32], 0, 0);
            __builtin_prefetch(&Bw[(size_t)(colBase + (tid >> 1)) * K + k0 + TILE_K + (tid & 1) * 32], 0, 0);
        }
#endif
        const bf16_t* Asl = &As[cur][0];
        const bf16_t* Bsl = &Bs[cur][0];

        // Two WMMA K=32 substeps over the staged 64-wide tile.
        #pragma unroll
        for (int kk = 0; kk < TILE_K; kk += TKK) {
            // A fragments (ISA 7.12.2, 16-bit A 16x32):
            // lane<16: K 0-7 & 16-23 of row (lane&15); lane>=16: K 8-15 & 24-31
            v16bf Af[2], Bf[4];
            #pragma unroll
            for (int mi = 0; mi < 2; ++mi) {
                int r = wm * 32 + mi * 16 + rsel;
                const bf16_t* p = &Asl[r * LDS_STR + kk + hsel * 8];
                #pragma unroll
                for (int e = 0; e < 8; ++e) { Af[mi][e] = p[e]; Af[mi][8 + e] = p[16 + e]; }
            }
            // B fragments: lane<16 holds K 0-7 & 16-23 of column (lane&15); lane>=16 K 8-15 & 24-31
            #pragma unroll
            for (int ni = 0; ni < 4; ++ni) {
                int n = wn * 64 + ni * 16 + rsel;
                const bf16_t* p = &Bsl[n * LDS_STR + kk + hsel * 8];
                #pragma unroll
                for (int e = 0; e < 8; ++e) { Bf[ni][e] = p[e]; Bf[ni][8 + e] = p[16 + e]; }
            }
            #pragma unroll
            for (int mi = 0; mi < 2; ++mi)
                #pragma unroll
                for (int ni = 0; ni < 4; ++ni)
                    acc[mi][ni] = __builtin_amdgcn_wmma_f32_16x16x32_bf16(
                        false, Af[mi], false, Bf[ni], (short)0, acc[mi][ni], false, false);
        }
        __syncthreads();                // all reads done before buffer reuse
#if HAVE_TDM
        cur ^= 1;
#endif
    }

    // C layout: VGPR j -> M = j + 8*(lane>>4); N = lane&15
    #pragma unroll
    for (int mi = 0; mi < 2; ++mi)
        #pragma unroll
        for (int ni = 0; ni < 4; ++ni) {
            int col = colBase + wn * 64 + ni * 16 + rsel;
            float bv = bias ? bias[col] : 0.0f;
            #pragma unroll
            for (int j = 0; j < 8; ++j) {
                int row = rowBase + wm * 32 + mi * 16 + hsel * 8 + j;
                C[(size_t)row * N + col] = acc[mi][ni][j] + bv;
            }
        }
}

// ---------------- SiLU + LRPE ----------------
__global__ __launch_bounds__(256)
void lrpe_kernel(const float* __restrict__ qkvu, const float* __restrict__ theta,
                 bf16_t* __restrict__ qL, bf16_t* __restrict__ kL) {
    int row = blockIdx.x;                    // b*NSEQ + n
    int b = row / NSEQ, n = row % NSEQ;
    int tid = threadIdx.x;
    #pragma unroll
    for (int i = 0; i < 4; ++i) {
        int d = tid + i * 256;               // 0..1023
        int h = d >> 6, dh = d & 63;
        float qv = silu_f(qkvu[(size_t)row * (4 * DMODEL) + d]);
        float kv = silu_f(qkvu[(size_t)row * (4 * DMODEL) + DMODEL + d]);
        float th = theta[d] * (float)n;
        float s, c;
        __sincosf(th, &s, &c);
        size_t base = ((size_t)(b * NHEAD + h) * NSEQ + n) * DLRPE;
        qL[base + dh]         = f2bf(qv * c);
        qL[base + DHEAD + dh] = f2bf(qv * s);
        kL[base + dh]         = f2bf(kv * c);
        kL[base + DHEAD + dh] = f2bf(kv * s);
    }
}

// ---------------- kv = kL^T @ v  (right-form linear attention) ----------------
__global__ __launch_bounds__(256)
void kv_kernel(const bf16_t* __restrict__ kL, const float* __restrict__ qkvu,
               bf16_t* __restrict__ kvT) {
    int z = blockIdx.x;                      // b*H + h
    int b = z / NHEAD, h = z % NHEAD;
    int tid = threadIdx.x;
    int d = tid >> 1;                        // 0..127
    int ebase = (tid & 1) * 32;              // 0 or 32

    __shared__ float vs[16][DHEAD];
    __shared__ float ks[16][DLRPE];

    float acc[32];
    #pragma unroll
    for (int e = 0; e < 32; ++e) acc[e] = 0.0f;

    for (int n0 = 0; n0 < NSEQ; n0 += 16) {
        #pragma unroll
        for (int i = 0; i < 4; ++i) {
            int idx = tid + i * 256;
            int r = idx >> 6, c = idx & 63;
            vs[r][c] = qkvu[((size_t)(b * NSEQ + n0 + r)) * (4 * DMODEL) + 2 * DMODEL + h * DHEAD + c];
        }
        #pragma unroll
        for (int i = 0; i < 8; ++i) {
            int idx = tid + i * 256;
            int r = idx >> 7, c = idx & 127;
            ks[r][c] = bf2f(kL[((size_t)z * NSEQ + n0 + r) * DLRPE + c]);
        }
        __syncthreads();
        #pragma unroll
        for (int r = 0; r < 16; ++r) {
            float kd = ks[r][d];
            #pragma unroll
            for (int e = 0; e < 32; ++e) acc[e] += kd * vs[r][ebase + e];
        }
        __syncthreads();
    }
    #pragma unroll
    for (int e = 0; e < 32; ++e)
        kvT[(size_t)z * (DHEAD * DLRPE) + (size_t)(ebase + e) * DLRPE + d] = f2bf(acc[e]);
}

// ---------------- per-head out = qL @ kv  (WMMA) ----------------
__global__ __launch_bounds__(256)
void gemm_head_wmma(const bf16_t* __restrict__ qL, const bf16_t* __restrict__ kvT,
                    float* __restrict__ outH) {
    __shared__ __align__(16) bf16_t As[128 * LDS_STR_H];
    __shared__ __align__(16) bf16_t Bs[64 * LDS_STR_H];

    const int z = blockIdx.z;
    const bf16_t* A  = qL  + (size_t)z * NSEQ * DLRPE;
    const bf16_t* Bw = kvT + (size_t)z * DHEAD * DLRPE;
    float* C = outH + (size_t)z * NSEQ * DHEAD;

    const int tid  = threadIdx.x;
    const int wave = tid >> 5;
    const int lane = tid & 31;
    const int wm   = wave >> 1;
    const int wn   = wave & 1;
    const int rowBase = blockIdx.y * 128;
    const int rsel = lane & 15;
    const int hsel = lane >> 4;

    v8f acc[2][2];
    #pragma unroll
    for (int mi = 0; mi < 2; ++mi)
        #pragma unroll
        for (int ni = 0; ni < 2; ++ni)
            #pragma unroll
            for (int j = 0; j < 8; ++j) acc[mi][ni][j] = 0.0f;

    for (int k0 = 0; k0 < DLRPE; k0 += TKK) {
        #pragma unroll
        for (int i = 0; i < 2; ++i) {        // A: 128x32
            int idx = tid + i * 256;
            int r = idx >> 2, c = (idx & 3) * 8;
            *(uint4*)&As[r * LDS_STR_H + c] =
                *(const uint4*)&A[(size_t)(rowBase + r) * DLRPE + k0 + c];
        }
        {                                     // B: 64x32
            int r = tid >> 2, c = (tid & 3) * 8;
            *(uint4*)&Bs[r * LDS_STR_H + c] =
                *(const uint4*)&Bw[(size_t)r * DLRPE + k0 + c];
        }
        __syncthreads();

        v16bf Af[2], Bf[2];
        #pragma unroll
        for (int mi = 0; mi < 2; ++mi) {
            int r = wm * 32 + mi * 16 + rsel;
            const bf16_t* p = &As[r * LDS_STR_H + hsel * 8];
            #pragma unroll
            for (int e = 0; e < 8; ++e) { Af[mi][e] = p[e]; Af[mi][8 + e] = p[16 + e]; }
        }
        #pragma unroll
        for (int ni = 0; ni < 2; ++ni) {
            int n = wn * 32 + ni * 16 + rsel;
            const bf16_t* p = &Bs[n * LDS_STR_H + hsel * 8];
            #pragma unroll
            for (int e = 0; e < 8; ++e) { Bf[ni][e] = p[e]; Bf[ni][8 + e] = p[16 + e]; }
        }
        #pragma unroll
        for (int mi = 0; mi < 2; ++mi)
            #pragma unroll
            for (int ni = 0; ni < 2; ++ni)
                acc[mi][ni] = __builtin_amdgcn_wmma_f32_16x16x32_bf16(
                    false, Af[mi], false, Bf[ni], (short)0, acc[mi][ni], false, false);
        __syncthreads();
    }

    #pragma unroll
    for (int mi = 0; mi < 2; ++mi)
        #pragma unroll
        for (int ni = 0; ni < 2; ++ni) {
            int col = wn * 32 + ni * 16 + rsel;
            #pragma unroll
            for (int j = 0; j < 8; ++j) {
                int row = rowBase + wm * 32 + mi * 16 + hsel * 8 + j;
                C[(size_t)row * DHEAD + col] = acc[mi][ni][j];
            }
        }
}

// ---------------- LayerNorm over D, multiply by u, emit bf16 ----------------
__global__ __launch_bounds__(256)
void ln_u_kernel(const float* __restrict__ outH, const float* __restrict__ qkvu,
                 const float* __restrict__ ln_w, const float* __restrict__ ln_b,
                 bf16_t* __restrict__ ybf) {
    int row = blockIdx.x;                    // b*NSEQ + n
    int b = row / NSEQ, n = row % NSEQ;
    int tid = threadIdx.x;

    __shared__ float rowv[DMODEL];
    __shared__ float s_sum[256], s_sq[256];

    float lsum = 0.0f, lsq = 0.0f;
    #pragma unroll
    for (int i = 0; i < 4; ++i) {
        int d = tid + i * 256;
        int h = d >> 6, e = d & 63;
        float v = outH[((size_t)(b * NHEAD + h) * NSEQ + n) * DHEAD + e];
        rowv[d] = v;
        lsum += v;
        lsq  += v * v;
    }
    s_sum[tid] = lsum; s_sq[tid] = lsq;
    __syncthreads();
    for (int s = 128; s > 0; s >>= 1) {
        if (tid < s) { s_sum[tid] += s_sum[tid + s]; s_sq[tid] += s_sq[tid + s]; }
        __syncthreads();
    }
    float mu  = s_sum[0] * (1.0f / DMODEL);
    float var = s_sq[0] * (1.0f / DMODEL) - mu * mu;
    float rinv = rsqrtf(var + LN_EPS);

    #pragma unroll
    for (int i = 0; i < 4; ++i) {
        int d = tid + i * 256;
        float y = (rowv[d] - mu) * rinv * ln_w[d] + ln_b[d];
        float u = qkvu[(size_t)row * (4 * DMODEL) + 3 * DMODEL + d];
        ybf[(size_t)row * DMODEL + d] = f2bf(y * u);
    }
}

// ---------------- launcher ----------------
extern "C" void kernel_launch(void* const* d_in, const int* in_sizes, int n_in,
                              void* d_out, int out_size, void* d_ws, size_t ws_size,
                              hipStream_t stream) {
    const float* x      = (const float*)d_in[0];
    // d_in[1] = context (unused by the reference math)
    const float* Wqkvu  = (const float*)d_in[2];
    const float* bqkvu  = (const float*)d_in[3];
    const float* Wout   = (const float*)d_in[4];
    const float* bout   = (const float*)d_in[5];
    const float* theta  = (const float*)d_in[6];
    const float* ln_w   = (const float*)d_in[7];
    const float* ln_b   = (const float*)d_in[8];
    float* out = (float*)d_out;

    char* ws = (char*)d_ws;
    size_t off = 0;
    auto carve = [&](size_t bytes) { char* p = ws + off; off = (off + bytes + 255) & ~(size_t)255; return p; };

    bf16_t* xbf     = (bf16_t*)carve((size_t)ROWS * DMODEL * 2);                 // 8 MB
    bf16_t* wqkvubf = (bf16_t*)carve((size_t)4 * DMODEL * DMODEL * 2);           // 8 MB
    bf16_t* woutbf  = (bf16_t*)carve((size_t)DMODEL * DMODEL * 2);               // 2 MB
    float*  qkvu    = (float*) carve((size_t)ROWS * 4 * DMODEL * 4);             // 64 MB
    bf16_t* qL      = (bf16_t*)carve((size_t)BATCH * NHEAD * NSEQ * DLRPE * 2);  // 16 MB
    bf16_t* kL      = (bf16_t*)carve((size_t)BATCH * NHEAD * NSEQ * DLRPE * 2);  // 16 MB
    bf16_t* kvT     = (bf16_t*)carve((size_t)BATCH * NHEAD * DHEAD * DLRPE * 2); // 0.5 MB
    float*  outH    = (float*) carve((size_t)BATCH * NHEAD * NSEQ * DHEAD * 4);  // 16 MB
    bf16_t* ybf     = (bf16_t*)carve((size_t)ROWS * DMODEL * 2);                 // 8 MB
    (void)ws_size; (void)in_sizes; (void)n_in; (void)out_size;

    cvt_f32_bf16<<<(ROWS * DMODEL + 255) / 256, 256, 0, stream>>>(x, xbf, ROWS * DMODEL);
    cvt_f32_bf16<<<(4 * DMODEL * DMODEL + 255) / 256, 256, 0, stream>>>(Wqkvu, wqkvubf, 4 * DMODEL * DMODEL);
    cvt_f32_bf16<<<(DMODEL * DMODEL + 255) / 256, 256, 0, stream>>>(Wout, woutbf, DMODEL * DMODEL);

    {   // qkvu = x @ Wqkvu^T + b
        dim3 g((4 * DMODEL) / TKN, ROWS / TKM);
        gemm_bf16_wmma<<<g, 256, 0, stream>>>(xbf, wqkvubf, bqkvu, qkvu, ROWS, 4 * DMODEL, DMODEL);
    }

    lrpe_kernel<<<ROWS, 256, 0, stream>>>(qkvu, theta, qL, kL);

    kv_kernel<<<BATCH * NHEAD, 256, 0, stream>>>(kL, qkvu, kvT);

    {   // out_heads = qL @ kv per head
        dim3 g(1, NSEQ / 128, BATCH * NHEAD);
        gemm_head_wmma<<<g, 256, 0, stream>>>(qL, kvT, outH);
    }

    ln_u_kernel<<<ROWS, 256, 0, stream>>>(outH, qkvu, ln_w, ln_b, ybf);

    {   // out = ybf @ Wout^T + bout
        dim3 g(DMODEL / TKN, ROWS / TKM);
        gemm_bf16_wmma<<<g, 256, 0, stream>>>(ybf, woutbf, bout, out, ROWS, DMODEL, DMODEL);
    }
}
